// RGN_5119601017229
// MI455X (gfx1250) — matrix operand order repaired and checked by reading, not compiled, not run on hardware
//
#include <hip/hip_runtime.h>
#include <hip/hip_bf16.h>
#include <math.h>

typedef _Float16 half_t;
typedef __attribute__((ext_vector_type(16))) _Float16 v16h;
typedef __attribute__((ext_vector_type(8)))  _Float16 v8h;
typedef __attribute__((ext_vector_type(8)))  float    v8f;

#define LSEQ    512
#define NB      64          // batch
#define HID     800
#define NGATES  3200        // 4*HID
#define ALPHA_N 60
#define NFRAG   22          // int(sqrt(512))
#define FRAGLEN 70          // ceil(1536/22)
#define T3      1536        // 3*LSEQ
#define NCHAIN  (NFRAG*NB)  // 1408
#define SCAN_BLOCKS 50      // 3200 gate rows / 64 per block

// ---------------------------------------------------------------------------
// WMMA fragment loaders (CDNA5 16-bit layouts, ISA 7.12.2)
// A 16x32 (MxK): lane<16 -> row=lane,   K = k0+{0..7, 16..23}
//                lane>=16-> row=lane-16,K = k0+{8..15,24..31}
// B 32x16 (KxN) from row-major W[n][k]: lane -> col = n0+(lane&15),
//                K = k0 + (lane>>4)*8 + {0..7, 16..23}  (two 16B loads)
// C/D 16x16 f32: lane&15 = N, vgpr v -> M = v + 8*(lane>>4)
// ---------------------------------------------------------------------------
__device__ inline v16h load_a(const half_t* __restrict__ A, int lda,
                              int m0, int k0, int lane) {
  int m  = m0 + (lane & 15);
  int kb = k0 + ((lane >> 4) << 3);
  const half_t* p = A + (size_t)m * lda + kb;
  v8h lo = *(const v8h*)p;
  v8h hi = *(const v8h*)(p + 16);
  v16h r;
#pragma unroll
  for (int i = 0; i < 8; ++i) { r[i] = lo[i]; r[i + 8] = hi[i]; }
  return r;
}

__device__ inline v16h load_b(const half_t* __restrict__ W, int ldw,
                              int n0, int k0, int lane) {
  int n  = n0 + (lane & 15);
  int kb = k0 + ((lane >> 4) << 3);
  const half_t* p = W + (size_t)n * ldw + kb;
  v8h lo = *(const v8h*)p;
  v8h hi = *(const v8h*)(p + 16);
  v16h r;
#pragma unroll
  for (int i = 0; i < 8; ++i) { r[i] = lo[i]; r[i + 8] = hi[i]; }
  return r;
}

__device__ inline float sigm(float x) { return 1.f / (1.f + __expf(-x)); }

// Lock-free grid barrier (persistent scan kernel): release add + acquire spin.
__device__ inline void grid_sync(unsigned* cnt, unsigned target) {
  __syncthreads();
  if (threadIdx.x == 0) {
    __threadfence();
    __hip_atomic_fetch_add(cnt, 1u, __ATOMIC_RELEASE, __HIP_MEMORY_SCOPE_AGENT);
    while (__hip_atomic_load(cnt, __ATOMIC_ACQUIRE, __HIP_MEMORY_SCOPE_AGENT) < target)
      __builtin_amdgcn_s_sleep(2);
    __threadfence();
  }
  __syncthreads();
}

// ---------------------------------------------------------------------------
// Input build: x16[l][b][64] = [one_hot(seq,20) | pssm(21) | zero pad to 64] fp16
// ---------------------------------------------------------------------------
__global__ void k_build_x16(const int* __restrict__ seq,
                            const float* __restrict__ pssm,
                            half_t* __restrict__ x16) {
  int r = blockIdx.x * blockDim.x + threadIdx.x;
  if (r >= LSEQ * NB) return;
  int s = seq[r];
  half_t* o = x16 + (size_t)r * 64;
#pragma unroll
  for (int k = 0; k < 20; ++k) o[k] = (half_t)(k == s ? 1.f : 0.f);
#pragma unroll
  for (int k = 0; k < 21; ++k) o[20 + k] = (half_t)pssm[(size_t)r * 21 + k];
#pragma unroll
  for (int k = 41; k < 64; ++k) o[k] = (half_t)0.f;
}

// Repack RNN weights: src [2][4*H][Ksrc] fp32 gate-major -> dst [2][3200][Kdst]
// fp16 with rows reordered to (j*4+gate) and K zero-padded to Kdst.
__global__ void k_prep_rnn_w(const float* __restrict__ src,
                             half_t* __restrict__ dst,
                             int Ksrc, int Kdst, int total) {
  int idx = blockIdx.x * blockDim.x + threadIdx.x;
  if (idx >= total) return;
  int k   = idx % Kdst;
  int rem = idx / Kdst;
  int row = rem % NGATES;
  int dir = rem / NGATES;
  int j = row >> 2, g = row & 3;
  float v = (k < Ksrc)
      ? src[((size_t)dir * NGATES + g * HID + j) * Ksrc + k] : 0.f;
  dst[(size_t)idx] = (half_t)v;
}

__global__ void k_prep_bias(const float* __restrict__ bih,
                            const float* __restrict__ bhh,
                            float* __restrict__ dst) {
  int idx = blockIdx.x * blockDim.x + threadIdx.x;
  if (idx >= 2 * NGATES) return;
  int row = idx % NGATES, dir = idx / NGATES;
  int j = row >> 2, g = row & 3;
  size_t s = (size_t)dir * NGATES + g * HID + j;
  dst[idx] = bih[s] + bhh[s];
}

__global__ void k_prep_fcw(const float* __restrict__ src, half_t* __restrict__ dst) {
  int idx = blockIdx.x * blockDim.x + threadIdx.x;
  if (idx >= 64 * 1600) return;
  int row = idx / 1600, k = idx % 1600;
  dst[idx] = (half_t)(row < ALPHA_N ? src[(size_t)row * 1600 + k] : 0.f);
}

// ---------------------------------------------------------------------------
// Persistent bidirectional-LSTM scan (one direction per launch).
// Grid: 50 blocks x 256 thr. 8 waves = 4 Mtiles x 2 N-groups; each wave owns
// TWO 16-wide N-tiles that share one A fragment (A reuse, fewer L2 reads).
// Block owns 64 gate rows = 16 hidden units x 4 gates (rows reordered j*4+g),
// so c-state lives in LDS for the whole sequence; h ping-pongs in global with
// one grid barrier per step. Bias is pre-splatted into the WMMA C operand.
// ---------------------------------------------------------------------------
template <int LDX>
__global__ __launch_bounds__(256) void k_lstm_scan(
    const half_t* __restrict__ xseq,   // [512][64][LDX]
    const half_t* __restrict__ Wih,    // [3200][LDX] reordered
    const half_t* __restrict__ Whh,    // [3200][800] reordered
    const float*  __restrict__ bias,   // [3200] reordered
    half_t* __restrict__ hs_out,       // [512][64][1600] (writes dir half)
    half_t* __restrict__ hbuf,         // 2 * 64*800 ping-pong
    unsigned* __restrict__ cnt, int dir) {
  __shared__ float gat[64][64];
  __shared__ float cst[64][16];
  const int tid  = threadIdx.x;
  const int lane = tid & 31, wave = tid >> 5;
  const int mt = wave & 3, nh = wave >> 2;       // nh in {0,1}: 32-col group
  const int nblk = blockIdx.x * 64;
  const int n0 = nblk + nh * 32;                 // first of two 16-col tiles
  const int ln = lane & 15, hi = lane >> 4;
  const float bv0 = bias[n0 + ln];               // constant across time
  const float bv1 = bias[n0 + 16 + ln];
  unsigned phase = 0;

  // init owned h columns of both ping-pong buffers + c-state
  for (int q = tid; q < 64 * 16; q += 256) {
    int m = q >> 4, jl = q & 15;
    int jg = (blockIdx.x << 4) + jl;
    hbuf[(size_t)m * HID + jg]            = (half_t)0.f;
    hbuf[(size_t)64 * HID + m * HID + jg] = (half_t)0.f;
    cst[m][jl] = 0.f;
  }
  phase += gridDim.x;
  grid_sync(cnt, phase);

  for (int s = 0; s < LSEQ; ++s) {
    const int tt = dir ? (LSEQ - 1 - s) : s;
    const half_t* xt = xseq + (size_t)tt * NB * LDX;
    const half_t* hp = hbuf + (size_t)(s & 1) * NB * HID;
    v8f acc0, acc1;
#pragma unroll
    for (int v = 0; v < 8; ++v) { acc0[v] = bv0; acc1[v] = bv1; }
#pragma unroll 4
    for (int k0 = 0; k0 < LDX; k0 += 32) {        // input projection
      v16h a  = load_a(xt, LDX, mt * 16, k0, lane);
      v16h b0 = load_b(Wih, LDX, n0, k0, lane);
      v16h b1 = load_b(Wih, LDX, n0 + 16, k0, lane);
      acc0 = __builtin_amdgcn_wmma_f32_16x16x32_f16(false, a, false, b0,
                                                    (short)0, acc0, false, false);
      acc1 = __builtin_amdgcn_wmma_f32_16x16x32_f16(false, a, false, b1,
                                                    (short)0, acc1, false, false);
    }
#pragma unroll 4
    for (int k0 = 0; k0 < HID; k0 += 32) {        // recurrence
      v16h a  = load_a(hp, HID, mt * 16, k0, lane);
      v16h b0 = load_b(Whh, HID, n0, k0, lane);
      v16h b1 = load_b(Whh, HID, n0 + 16, k0, lane);
      acc0 = __builtin_amdgcn_wmma_f32_16x16x32_f16(false, a, false, b0,
                                                    (short)0, acc0, false, false);
      acc1 = __builtin_amdgcn_wmma_f32_16x16x32_f16(false, a, false, b1,
                                                    (short)0, acc1, false, false);
    }
    { // D tiles -> LDS (C/D layout: lane&15 = N, vgpr v -> M = v + 8*(lane>>4))
      int rbase = mt * 16 + hi * 8, cbase = nh * 32;
#pragma unroll
      for (int v = 0; v < 8; ++v) {
        gat[rbase + v][cbase + ln]      = acc0[v];
        gat[rbase + v][cbase + 16 + ln] = acc1[v];
      }
    }
    __syncthreads();

    half_t* hn = hbuf + (size_t)((s + 1) & 1) * NB * HID;
    for (int q = tid; q < 64 * 16; q += 256) {
      int m = q >> 4, jl = q & 15, cb = jl * 4;
      float gi = gat[m][cb + 0];
      float gf = gat[m][cb + 1];
      float gg = gat[m][cb + 2];
      float go = gat[m][cb + 3];
      float cn = sigm(gf) * cst[m][jl] + sigm(gi) * tanhf(gg);
      cst[m][jl] = cn;
      float h = sigm(go) * tanhf(cn);
      int jg = (blockIdx.x << 4) + jl;
      hn[(size_t)m * HID + jg] = (half_t)h;
      hs_out[(size_t)tt * NB * 1600 + (size_t)m * 1600 + dir * HID + jg] = (half_t)h;
    }
    phase += gridDim.x;
    grid_sync(cnt, phase);
  }
}

// ---------------------------------------------------------------------------
// FC GEMM: logits(32768x64) = hs1(32768x1600 f16) @ fcw16^T, WMMA tiles
// ---------------------------------------------------------------------------
__global__ __launch_bounds__(256) void k_fc_gemm(const half_t* __restrict__ A,
                                                 const half_t* __restrict__ Bw,
                                                 float* __restrict__ out) {
  int lane = threadIdx.x & 31, wave = threadIdx.x >> 5;
  int id = blockIdx.x * 8 + wave;
  int m0 = (id >> 2) * 16, n0 = (id & 3) * 16;
  v8f acc = {};
#pragma unroll 4
  for (int k0 = 0; k0 < 1600; k0 += 32) {
    v16h a = load_a(A, 1600, m0, k0, lane);
    v16h b = load_b(Bw, 1600, n0, k0, lane);
    acc = __builtin_amdgcn_wmma_f32_16x16x32_f16(false, a, false, b,
                                                 (short)0, acc, false, false);
  }
  int ln = lane & 15, hi = lane >> 4;
#pragma unroll
  for (int v = 0; v < 8; ++v)
    out[(size_t)(m0 + hi * 8 + v) * 64 + n0 + ln] = acc[v];
}

// softmax -> phi = atan2(p@sinA, p@cosA) -> SRF rows (3l+k)
__global__ void k_softmax_srf(const float* __restrict__ logits,
                              const float* __restrict__ fcb,
                              const float* __restrict__ alphabet,
                              const float* __restrict__ bl,
                              const float* __restrict__ ba,
                              float* __restrict__ srf) {
  int r = blockIdx.x * blockDim.x + threadIdx.x;
  if (r >= LSEQ * NB) return;
  float p[ALPHA_N];
  float mx = -1e30f;
  for (int a = 0; a < ALPHA_N; ++a) {
    p[a] = logits[(size_t)r * 64 + a] + fcb[a];
    mx = fmaxf(mx, p[a]);
  }
  float sum = 0.f;
  for (int a = 0; a < ALPHA_N; ++a) { p[a] = __expf(p[a] - mx); sum += p[a]; }
  float inv = 1.f / sum;
  int l = r >> 6, b = r & 63;
  for (int k = 0; k < 3; ++k) {
    float ss = 0.f, cc = 0.f;
    for (int a = 0; a < ALPHA_N; ++a) {
      float ang = alphabet[a * 3 + k];
      ss += p[a] * __sinf(ang);
      cc += p[a] * __cosf(ang);
    }
    float phi = atan2f(ss * inv, cc * inv);
    float t  = 3.14159265358979323846f - ba[k];
    float rc = bl[k] * __cosf(t);
    float rs = bl[k] * __sinf(t);
    float* o = srf + (size_t)(3 * l + k) * (NB * 3) + b * 3;
    o[0] = rc; o[1] = rs * __cosf(phi); o[2] = rs * __sinf(phi);
  }
}

// pNeRF extend scan: 1408 independent chains x 70 steps
__global__ void k_pnerf_extend(const float* __restrict__ srf,
                               float* __restrict__ pts) {
  int ch = blockIdx.x * blockDim.x + threadIdx.x;
  if (ch >= NCHAIN) return;
  int f = ch >> 6, b = ch & 63;
  float A[3] = {-0.70710678f, 1.22474487f, 0.f};
  float B[3] = {-1.41421356f, 0.f, 0.f};
  float C[3] = {0.f, 0.f, 0.f};
  for (int p = 0; p < FRAGLEN; ++p) {
    int row = f * FRAGLEN + p;
    float d0 = 0.f, d1 = 0.f, d2 = 0.f;
    if (row < T3) {
      const float* s = srf + (size_t)row * (NB * 3) + b * 3;
      d0 = s[0]; d1 = s[1]; d2 = s[2];
    }
    float bc[3] = {C[0] - B[0], C[1] - B[1], C[2] - B[2]};
    float nb = sqrtf(bc[0]*bc[0] + bc[1]*bc[1] + bc[2]*bc[2]) + 1e-12f;
    bc[0] /= nb; bc[1] /= nb; bc[2] /= nb;
    float ab[3] = {B[0] - A[0], B[1] - A[1], B[2] - A[2]};
    float n3[3] = {ab[1]*bc[2] - ab[2]*bc[1],
                   ab[2]*bc[0] - ab[0]*bc[2],
                   ab[0]*bc[1] - ab[1]*bc[0]};
    float nn = sqrtf(n3[0]*n3[0] + n3[1]*n3[1] + n3[2]*n3[2]) + 1e-12f;
    n3[0] /= nn; n3[1] /= nn; n3[2] /= nn;
    float nx[3] = {n3[1]*bc[2] - n3[2]*bc[1],
                   n3[2]*bc[0] - n3[0]*bc[2],
                   n3[0]*bc[1] - n3[1]*bc[0]};
    float D[3];
    float* o = pts + (size_t)p * NCHAIN * 3 + (size_t)ch * 3;
#pragma unroll
    for (int j = 0; j < 3; ++j) {
      D[j] = C[j] + d0 * bc[j] + d1 * nx[j] + d2 * n3[j];
      o[j] = D[j];
    }
#pragma unroll
    for (int j = 0; j < 3; ++j) { A[j] = B[j]; B[j] = C[j]; C[j] = D[j]; }
  }
}

// pNeRF compose scan over fragments (per batch element)
__global__ void k_pnerf_compose(const float* __restrict__ pts,
                                float* __restrict__ Gms,
                                float* __restrict__ Gts) {
  int b = threadIdx.x;
  if (b >= NB) return;
  float Gm[9] = {1,0,0, 0,1,0, 0,0,1};
  float Gt[3] = {0,0,0};
  for (int f = 0; f < NFRAG; ++f) {
    int ch = f * 64 + b;
    const float* pa = pts + (size_t)(FRAGLEN-3) * NCHAIN * 3 + (size_t)ch * 3;
    const float* pb = pts + (size_t)(FRAGLEN-2) * NCHAIN * 3 + (size_t)ch * 3;
    const float* pc = pts + (size_t)(FRAGLEN-1) * NCHAIN * 3 + (size_t)ch * 3;
#pragma unroll
    for (int i = 0; i < 9; ++i) Gms[(size_t)f * NB * 9 + b * 9 + i] = Gm[i];
#pragma unroll
    for (int j = 0; j < 3; ++j) Gts[(size_t)f * NB * 3 + b * 3 + j] = Gt[j];
    float bc[3] = {pc[0]-pb[0], pc[1]-pb[1], pc[2]-pb[2]};
    float nb2 = sqrtf(bc[0]*bc[0] + bc[1]*bc[1] + bc[2]*bc[2]) + 1e-12f;
    bc[0]/=nb2; bc[1]/=nb2; bc[2]/=nb2;
    float ab[3] = {pb[0]-pa[0], pb[1]-pa[1], pb[2]-pa[2]};
    float n3[3] = {ab[1]*bc[2]-ab[2]*bc[1], ab[2]*bc[0]-ab[0]*bc[2], ab[0]*bc[1]-ab[1]*bc[0]};
    float nn = sqrtf(n3[0]*n3[0] + n3[1]*n3[1] + n3[2]*n3[2]) + 1e-12f;
    n3[0]/=nn; n3[1]/=nn; n3[2]/=nn;
    float M[9] = {bc[0], bc[1], bc[2],
                  n3[1]*bc[2]-n3[2]*bc[1], n3[2]*bc[0]-n3[0]*bc[2], n3[0]*bc[1]-n3[1]*bc[0],
                  n3[0], n3[1], n3[2]};
    float nG[9], nT[3];
#pragma unroll
    for (int i = 0; i < 3; ++i)
#pragma unroll
      for (int j = 0; j < 3; ++j)
        nG[i*3+j] = M[i*3+0]*Gm[0*3+j] + M[i*3+1]*Gm[1*3+j] + M[i*3+2]*Gm[2*3+j];
#pragma unroll
    for (int j = 0; j < 3; ++j)
      nT[j] = Gt[j] + pc[0]*Gm[0*3+j] + pc[1]*Gm[1*3+j] + pc[2]*Gm[2*3+j];
#pragma unroll
    for (int i = 0; i < 9; ++i) Gm[i] = nG[i];
#pragma unroll
    for (int j = 0; j < 3; ++j) Gt[j] = nT[j];
  }
}

__global__ void k_pnerf_final(const float* __restrict__ pts,
                              const float* __restrict__ Gms,
                              const float* __restrict__ Gts,
                              float* __restrict__ out) {
  int idx = blockIdx.x * blockDim.x + threadIdx.x;
  if (idx >= T3 * NB) return;
  int row = idx >> 6, b = idx & 63;
  int f = row / FRAGLEN, p = row % FRAGLEN;
  const float* fr = pts + (size_t)p * NCHAIN * 3 + (size_t)(f * 64 + b) * 3;
  const float* G  = Gms + (size_t)f * NB * 9 + b * 9;
  const float* Tt = Gts + (size_t)f * NB * 3 + b * 3;
  float* o = out + (size_t)row * (NB * 3) + b * 3;
#pragma unroll
  for (int j = 0; j < 3; ++j)
    o[j] = Tt[j] + fr[0]*G[0*3+j] + fr[1]*G[1*3+j] + fr[2]*G[2*3+j];
}

// ---------------------------------------------------------------------------
extern "C" void kernel_launch(void* const* d_in, const int* in_sizes, int n_in,
                              void* d_out, int out_size, void* d_ws, size_t ws_size,
                              hipStream_t stream) {
  const int*   seq      = (const int*)d_in[0];
  const float* pssm     = (const float*)d_in[2];
  const float* w_ih_l0  = (const float*)d_in[4];
  const float* w_hh_l0  = (const float*)d_in[5];
  const float* b_ih_l0  = (const float*)d_in[6];
  const float* b_hh_l0  = (const float*)d_in[7];
  const float* w_ih_l1  = (const float*)d_in[8];
  const float* w_hh_l1  = (const float*)d_in[9];
  const float* b_ih_l1  = (const float*)d_in[10];
  const float* b_hh_l1  = (const float*)d_in[11];
  const float* fc_w     = (const float*)d_in[12];
  const float* fc_b     = (const float*)d_in[13];
  const float* alphabet = (const float*)d_in[14];
  const float* bl       = (const float*)d_in[15];
  const float* ba       = (const float*)d_in[16];

  char* ws = (char*)d_ws;
  size_t off = 0;
  auto alloc = [&](size_t bytes) -> char* {
    char* p = ws + off;
    off = (off + bytes + 255) & ~(size_t)255;
    return p;
  };
  half_t*   x16    = (half_t*)alloc((size_t)LSEQ * NB * 64 * 2);
  half_t*   w0ih   = (half_t*)alloc((size_t)2 * NGATES * 64 * 2);
  half_t*   w0hh   = (half_t*)alloc((size_t)2 * NGATES * HID * 2);
  half_t*   w1ih   = (half_t*)alloc((size_t)2 * NGATES * 1600 * 2);
  half_t*   w1hh   = (half_t*)alloc((size_t)2 * NGATES * HID * 2);
  float*    b0     = (float*)alloc((size_t)2 * NGATES * 4);
  float*    b1     = (float*)alloc((size_t)2 * NGATES * 4);
  half_t*   fcw16  = (half_t*)alloc((size_t)64 * 1600 * 2);
  half_t*   hs0    = (half_t*)alloc((size_t)LSEQ * NB * 1600 * 2);
  half_t*   hs1    = (half_t*)alloc((size_t)LSEQ * NB * 1600 * 2);
  half_t*   hbuf   = (half_t*)alloc((size_t)2 * NB * HID * 2);
  float*    logits = (float*)alloc((size_t)LSEQ * NB * 64 * 4);
  float*    srf    = (float*)alloc((size_t)T3 * NB * 3 * 4);
  float*    pts    = (float*)alloc((size_t)FRAGLEN * NCHAIN * 3 * 4);
  float*    gms    = (float*)alloc((size_t)NFRAG * NB * 9 * 4);
  float*    gts    = (float*)alloc((size_t)NFRAG * NB * 3 * 4);
  unsigned* cnts   = (unsigned*)alloc(256);

  hipMemsetAsync(cnts, 0, 256, stream);

  // precision prep (all L2-resident afterwards: ~42MB fp16 weights)
  k_build_x16<<<(LSEQ * NB + 255) / 256, 256, 0, stream>>>(seq, pssm, x16);
  k_prep_rnn_w<<<(2 * NGATES * 64 + 255) / 256, 256, 0, stream>>>(
      w_ih_l0, w0ih, 41, 64, 2 * NGATES * 64);
  k_prep_rnn_w<<<(2 * NGATES * HID + 255) / 256, 256, 0, stream>>>(
      w_hh_l0, w0hh, HID, HID, 2 * NGATES * HID);
  k_prep_rnn_w<<<(2 * NGATES * 1600 + 255) / 256, 256, 0, stream>>>(
      w_ih_l1, w1ih, 1600, 1600, 2 * NGATES * 1600);
  k_prep_rnn_w<<<(2 * NGATES * HID + 255) / 256, 256, 0, stream>>>(
      w_hh_l1, w1hh, HID, HID, 2 * NGATES * HID);
  k_prep_bias<<<(2 * NGATES + 255) / 256, 256, 0, stream>>>(b_ih_l0, b_hh_l0, b0);
  k_prep_bias<<<(2 * NGATES + 255) / 256, 256, 0, stream>>>(b_ih_l1, b_hh_l1, b1);
  k_prep_fcw<<<(64 * 1600 + 255) / 256, 256, 0, stream>>>(fc_w, fcw16);

  // layer 0 (forward then backward), then layer 1 on the concatenated output
  k_lstm_scan<64><<<SCAN_BLOCKS, 256, 0, stream>>>(
      x16, w0ih, w0hh, b0, hs0, hbuf, cnts + 0, 0);
  k_lstm_scan<64><<<SCAN_BLOCKS, 256, 0, stream>>>(
      x16, w0ih + (size_t)NGATES * 64, w0hh + (size_t)NGATES * HID,
      b0 + NGATES, hs0, hbuf, cnts + 16, 1);
  k_lstm_scan<1600><<<SCAN_BLOCKS, 256, 0, stream>>>(
      hs0, w1ih, w1hh, b1, hs1, hbuf, cnts + 32, 0);
  k_lstm_scan<1600><<<SCAN_BLOCKS, 256, 0, stream>>>(
      hs0, w1ih + (size_t)NGATES * 1600, w1hh + (size_t)NGATES * HID,
      b1 + NGATES, hs1, hbuf, cnts + 48, 1);

  // head + geometry
  k_fc_gemm<<<1024, 256, 0, stream>>>(hs1, fcw16, logits);
  k_softmax_srf<<<(LSEQ * NB + 255) / 256, 256, 0, stream>>>(
      logits, fc_b, alphabet, bl, ba, srf);
  k_pnerf_extend<<<(NCHAIN + 255) / 256, 256, 0, stream>>>(srf, pts);
  k_pnerf_compose<<<1, 64, 0, stream>>>(pts, gms, gts);
  k_pnerf_final<<<(T3 * NB + 255) / 256, 256, 0, stream>>>(
      pts, gms, gts, (float*)d_out);
}
